// NeuralSDF_42777874268460
// MI455X (gfx1250) — matrix operand order repaired and checked by reading, not compiled, not run on hardware
//
#include <hip/hip_runtime.h>
#include <hip/hip_bf16.h>

typedef __attribute__((ext_vector_type(16))) _Float16 v16h;
typedef __attribute__((ext_vector_type(8)))  float    v8f;

#define NPTS   524288
#define EMB    16
#define RES    128
#define HID    128
#define DINP   19
#define KPAD   32
#define SIRW   30.0f
#define WAVES  4
#define THREADS (WAVES * 32)

// ---- LDS layout (dynamic shared) ----
#define OFF_W0   0            // f16 [128][32]   W0[n][k] fwd-B          8192 B
#define OFF_W1   8192         // f16 [128][128]  W1[n][k] fwd-B         32768 B
#define OFF_W1T  40960        // f16 [128][128]  W1T[n][k]=W1[k][n] bwd 32768 B
#define OFF_W0T  73728        // f16 [32][128]   W0T[n][k]=W0[k][n] bwd  8192 B
#define OFF_B0   81920        // f32 [128]
#define OFF_B1   82432        // f32 [128]
#define OFF_WO   82944        // f32 [128]
#define OFF_WAVE 83456
#define WV_XS    0            // f16 [16][32]
#define WV_H0    1024         // f16 [16][128]
#define WV_C0    5120         // f16 [16][128]
#define WV_GS    9216         // f16 [16][128]
#define WV_GE    13312        // f32 [16][16]
#define WV_GP    14336        // f32 [16][4]
#define WV_SZ    14592
#define SMEM_BYTES (OFF_WAVE + WAVES * WV_SZ)   // 141824

// A-matrix (16-bit, 16x32) K index for vgpr i, lane-half hi (ISA 7.12.2)
__device__ __forceinline__ int kb_A(int i, int hi) {
    return (i < 4 ? 2 * i : 2 * i + 8) + hi * 8;
}

// A fragment from row-major LDS tile: value = base[row*ld + k]
// (halves cover k = hi*8..hi*8+7 and 16+hi*8..23+hi*8 -> two contiguous 16B runs)
__device__ __forceinline__ v16h frag_A(const _Float16* base, int ld, int row, int k0, int hi) {
    v16h r;
#pragma unroll
    for (int t = 0; t < 16; ++t) {
        int i = t >> 1;
        r[t] = base[row * ld + k0 + kb_A(i, hi) + (t & 1)];
    }
    return r;
}

// B fragment (32x16): lane half hi holds K = k0 + 8*hi + t (t<8), k0 + 16 + 8*hi + (t-8) (t>=8), column n.
// Storage row-major over the OUTPUT index n with k contiguous: value B[k][n] = W[n*ld + k]
// -> two contiguous 16B runs per lane (two ds_load_b128).
__device__ __forceinline__ v16h frag_B(const _Float16* W, int ld, int n, int k0, int hi) {
    v16h r;
#pragma unroll
    for (int t = 0; t < 16; ++t)
        r[t] = W[n * ld + k0 + (t < 8 ? t : t + 8) + hi * 8];
    return r;
}

__device__ __forceinline__ v8f wmma_f16(v16h a, v16h b, v8f c) {
    return __builtin_amdgcn_wmma_f32_16x16x32_f16(false, a, false, b, (short)0, c, false, false);
}

__global__ __launch_bounds__(THREADS) void neural_sdf_fused(
    const float* __restrict__ pos, const float* __restrict__ gridp,
    const float* __restrict__ W0g, const float* __restrict__ b0g,
    const float* __restrict__ W1g, const float* __restrict__ b1g,
    const float* __restrict__ Wog, const float* __restrict__ bog,
    float* __restrict__ out)
{
    extern __shared__ char smem[];
    _Float16* w0s  = (_Float16*)(smem + OFF_W0);
    _Float16* w1s  = (_Float16*)(smem + OFF_W1);
    _Float16* w1ts = (_Float16*)(smem + OFF_W1T);
    _Float16* w0ts = (_Float16*)(smem + OFF_W0T);
    float* b0f = (float*)(smem + OFF_B0);
    float* b1f = (float*)(smem + OFF_B1);
    float* wof = (float*)(smem + OFF_WO);

    const int wave = threadIdx.x >> 5;
    char* wb = smem + OFF_WAVE + wave * WV_SZ;
    _Float16* xs  = (_Float16*)(wb + WV_XS);
    _Float16* h0s = (_Float16*)(wb + WV_H0);
    _Float16* c0s = (_Float16*)(wb + WV_C0);
    _Float16* gs  = (_Float16*)(wb + WV_GS);
    float*    ges = (float*)(wb + WV_GE);
    float*    gps = (float*)(wb + WV_GP);

    // ---- cooperative weight staging (normal + transposed copies) ----
    for (int idx = threadIdx.x; idx < HID * KPAD; idx += THREADS) {
        int h = idx >> 5, k = idx & 31;
        w0s[idx] = (_Float16)(k < DINP ? W0g[h * DINP + k] : 0.0f);
    }
    for (int idx = threadIdx.x; idx < KPAD * HID; idx += THREADS) {
        int n = idx >> 7, k = idx & 127;          // w0ts[n][k] = W0[k][n]
        w0ts[idx] = (_Float16)(n < DINP ? W0g[k * DINP + n] : 0.0f);
    }
    for (int idx = threadIdx.x; idx < HID * HID; idx += THREADS) {
        w1s[idx] = (_Float16)W1g[idx];            // w1s[n][k] = W1[n][k] (fwd: B=W1^T)
        int n = idx >> 7, k = idx & 127;
        w1ts[idx] = (_Float16)W1g[k * HID + n];   // w1ts[n][k] = W1[k][n] (bwd: B=W1)
    }
    for (int idx = threadIdx.x; idx < HID; idx += THREADS) {
        b0f[idx] = b0g[idx];
        b1f[idx] = b1g[idx];
        wof[idx] = Wog[idx];
    }
    __syncthreads();
    const float bof = bog[0];

    const int lane = threadIdx.x & 31;
    const int hi = lane >> 4;      // lane half: A/B K-split, C row-split
    const int c  = lane & 15;      // point-in-tile / matrix column
    const int fh = hi;             // feature-half for the gather

    const int nTiles = NPTS / 16;
    for (int tile = blockIdx.x * WAVES + wave; tile < nTiles; tile += gridDim.x * WAVES) {
        const int ptBase = tile * 16;
        const int pt = ptBase + c;

        // ================= gather pass 1: trilinear embedding =================
        float px = pos[pt * 3 + 0] * 0.5f + 0.5f;
        float py = pos[pt * 3 + 1] * 0.5f + 0.5f;
        float pz = pos[pt * 3 + 2] * 0.5f + 0.5f;
        float xg = px * (float)(RES - 1);
        float yg = py * (float)(RES - 1);
        float zg = pz * (float)(RES - 1);
        float fix = fminf(fmaxf(floorf(xg), 0.0f), (float)(RES - 2));
        float fiy = fminf(fmaxf(floorf(yg), 0.0f), (float)(RES - 2));
        float fiz = fminf(fmaxf(floorf(zg), 0.0f), (float)(RES - 2));
        int ix = (int)fix, iy = (int)fiy, iz = (int)fiz;
        float fx = xg - fix, fy = yg - fiy, fz = zg - fiz;

        float emb[8] = {0, 0, 0, 0, 0, 0, 0, 0};
#pragma unroll
        for (int cr = 0; cr < 8; ++cr) {
            int dx = cr >> 2, dy = (cr >> 1) & 1, dz = cr & 1;
            float w = (dx ? fx : 1.0f - fx) * (dy ? fy : 1.0f - fy) * (dz ? fz : 1.0f - fz);
            const float* gp = gridp + (size_t)(((ix + dx) * RES + (iy + dy)) * RES + (iz + dz)) * EMB + fh * 8;
            float4 g0 = *(const float4*)(gp + 0);
            float4 g1 = *(const float4*)(gp + 4);
            emb[0] += w * g0.x; emb[1] += w * g0.y; emb[2] += w * g0.z; emb[3] += w * g0.w;
            emb[4] += w * g1.x; emb[5] += w * g1.y; emb[6] += w * g1.z; emb[7] += w * g1.w;
        }
#pragma unroll
        for (int j = 0; j < 8; ++j)
            xs[c * KPAD + fh * 8 + j] = (_Float16)emb[j];
        if (hi == 0) {
            xs[c * KPAD + 16] = (_Float16)px;
            xs[c * KPAD + 17] = (_Float16)py;
            xs[c * KPAD + 18] = (_Float16)pz;
#pragma unroll
            for (int k = 19; k < 32; ++k) xs[c * KPAD + k] = (_Float16)0.0f;
        }
        __builtin_amdgcn_wave_barrier();

        // ================= layer 0: z0 = X @ W0^T  (8 WMMAs) =================
        v16h a0 = frag_A(xs, KPAD, c, 0, hi);
        v8f z0acc[8];
#pragma unroll
        for (int nt = 0; nt < 8; ++nt) {
            v8f cz = {0, 0, 0, 0, 0, 0, 0, 0};
            v16h b = frag_B(w0s, KPAD, nt * 16 + c, 0, hi);   // B[k][n] = W0[n][k]
            z0acc[nt] = wmma_f16(a0, b, cz);
        }
        // sin / 30cos, staged f16 for layer1 A and backward elementwise
#pragma unroll
        for (int nt = 0; nt < 8; ++nt) {
            int n = nt * 16 + c;
            float bb = b0f[n];
#pragma unroll
            for (int j = 0; j < 8; ++j) {
                int m = j + hi * 8;
                float z = z0acc[nt][j] + bb;
                h0s[m * HID + n] = (_Float16)__sinf(SIRW * z);
                c0s[m * HID + n] = (_Float16)(SIRW * __cosf(SIRW * z));
            }
        }
        __builtin_amdgcn_wave_barrier();

        // ================= layer 1: z1 = h0 @ W1^T  (32 WMMAs) =================
        v8f z1acc[8];
#pragma unroll
        for (int nt = 0; nt < 8; ++nt) {
            v8f cz = {0, 0, 0, 0, 0, 0, 0, 0};
            z1acc[nt] = cz;
        }
#pragma unroll
        for (int kk = 0; kk < 4; ++kk) {
            v16h a = frag_A(h0s, HID, c, kk * 32, hi);
#pragma unroll
            for (int nt = 0; nt < 8; ++nt) {
                v16h b = frag_B(w1s, HID, nt * 16 + c, kk * 32, hi);  // W1[n][k]
                z1acc[nt] = wmma_f16(a, b, z1acc[nt]);
            }
        }

        // ===== output dot + g1 = Wo * 30cos(30 z1), staged to gs =====
        float sdfp[8] = {0, 0, 0, 0, 0, 0, 0, 0};
#pragma unroll
        for (int nt = 0; nt < 8; ++nt) {
            int n = nt * 16 + c;
            float wo = wof[n];
            float bb = b1f[n];
#pragma unroll
            for (int j = 0; j < 8; ++j) {
                float z = z1acc[nt][j] + bb;
                sdfp[j] += wo * __sinf(SIRW * z);
                gs[(j + hi * 8) * HID + n] = (_Float16)(wo * SIRW * __cosf(SIRW * z));
            }
        }
#pragma unroll
        for (int j = 0; j < 8; ++j) {
            float v = sdfp[j];
            v += __shfl_xor(v, 1, 32);
            v += __shfl_xor(v, 2, 32);
            v += __shfl_xor(v, 4, 32);
            v += __shfl_xor(v, 8, 32);
            sdfp[j] = v;
        }
        if (c == 0) {
#pragma unroll
            for (int j = 0; j < 8; ++j)
                out[ptBase + j + hi * 8] = sdfp[j] + bof;
        }
        __builtin_amdgcn_wave_barrier();

        // ================= backward 1: g0 = (g1 @ W1) * c0  (32 WMMAs) =================
        v8f g0acc[8];
#pragma unroll
        for (int nt = 0; nt < 8; ++nt) {
            v8f cz = {0, 0, 0, 0, 0, 0, 0, 0};
            g0acc[nt] = cz;
        }
#pragma unroll
        for (int kk = 0; kk < 4; ++kk) {
            v16h a = frag_A(gs, HID, c, kk * 32, hi);
#pragma unroll
            for (int nt = 0; nt < 8; ++nt) {
                v16h b = frag_B(w1ts, HID, nt * 16 + c, kk * 32, hi);  // B[k][n] = W1[k][n]
                g0acc[nt] = wmma_f16(a, b, g0acc[nt]);
            }
        }
#pragma unroll
        for (int nt = 0; nt < 8; ++nt) {
            int n = nt * 16 + c;
#pragma unroll
            for (int j = 0; j < 8; ++j) {
                int m = j + hi * 8;
                float g = g0acc[nt][j] * (float)c0s[m * HID + n];
                gs[m * HID + n] = (_Float16)g;   // overwrite g1 (all reads above done, same wave)
            }
        }
        __builtin_amdgcn_wave_barrier();

        // ============ backward 2: gin = g0 @ W0  (8 WMMAs; cols 0-15 emb, 16-18 p) ============
        v8f geacc = {0, 0, 0, 0, 0, 0, 0, 0};
        v8f gpacc = {0, 0, 0, 0, 0, 0, 0, 0};
#pragma unroll
        for (int kk = 0; kk < 4; ++kk) {
            v16h a = frag_A(gs, HID, c, kk * 32, hi);
            v16h be = frag_B(w0ts, HID, c, kk * 32, hi);        // B[k][n]=W0[k][n], n=0..15
            v16h bp = frag_B(w0ts, HID, 16 + c, kk * 32, hi);   // n=16..31 (>=19 are zero pad)
            geacc = wmma_f16(a, be, geacc);
            gpacc = wmma_f16(a, bp, gpacc);
        }
#pragma unroll
        for (int j = 0; j < 8; ++j) {
            int m = j + hi * 8;
            ges[m * 16 + c] = geacc[j];
        }
        if (c < 3) {
#pragma unroll
            for (int j = 0; j < 8; ++j)
                gps[(j + hi * 8) * 4 + c] = gpacc[j];
        }
        __builtin_amdgcn_wave_barrier();

        // ============ gather pass 2: grad_p += 127 * J_trilinear^T @ grad_emb ============
        float gx = 0.0f, gy = 0.0f, gz = 0.0f;
#pragma unroll
        for (int cr = 0; cr < 8; ++cr) {
            int dx = cr >> 2, dy = (cr >> 1) & 1, dz = cr & 1;
            const float* gp = gridp + (size_t)(((ix + dx) * RES + (iy + dy)) * RES + (iz + dz)) * EMB + fh * 8;
            float4 g0 = *(const float4*)(gp + 0);
            float4 g1 = *(const float4*)(gp + 4);
            const float* ge = ges + c * 16 + fh * 8;
            float s = g0.x * ge[0] + g0.y * ge[1] + g0.z * ge[2] + g0.w * ge[3]
                    + g1.x * ge[4] + g1.y * ge[5] + g1.z * ge[6] + g1.w * ge[7];
            float wx = dx ? fx : 1.0f - fx, wy = dy ? fy : 1.0f - fy, wz = dz ? fz : 1.0f - fz;
            float sx = dx ? 1.0f : -1.0f, sy = dy ? 1.0f : -1.0f, sz = dz ? 1.0f : -1.0f;
            gx += sx * wy * wz * s;
            gy += wx * sy * wz * s;
            gz += wx * wy * sz * s;
        }
        gx *= (float)(RES - 1);
        gy *= (float)(RES - 1);
        gz *= (float)(RES - 1);
        gx += __shfl_xor(gx, 16, 32);
        gy += __shfl_xor(gy, 16, 32);
        gz += __shfl_xor(gz, 16, 32);
        if (hi == 0) {
            out[NPTS + pt * 3 + 0] = gx + gps[c * 4 + 0];
            out[NPTS + pt * 3 + 1] = gy + gps[c * 4 + 1];
            out[NPTS + pt * 3 + 2] = gz + gps[c * 4 + 2];
        }
        __builtin_amdgcn_wave_barrier();
    }
}

extern "C" void kernel_launch(void* const* d_in, const int* in_sizes, int n_in,
                              void* d_out, int out_size, void* d_ws, size_t ws_size,
                              hipStream_t stream) {
    const float* pos = (const float*)d_in[0];
    const float* grd = (const float*)d_in[1];
    const float* W0  = (const float*)d_in[2];
    const float* b0  = (const float*)d_in[3];
    const float* W1  = (const float*)d_in[4];
    const float* b1  = (const float*)d_in[5];
    const float* Wo  = (const float*)d_in[6];
    const float* bo  = (const float*)d_in[7];
    float* out = (float*)d_out;

    // 32768 tiles of 16 points; 8192 blocks x 4 waves = one tile per wave.
    dim3 blocks(8192), threads(THREADS);
    neural_sdf_fused<<<blocks, threads, SMEM_BYTES, stream>>>(
        pos, grd, W0, b0, W1, b1, Wo, bo, out);
}
